// JistModelMe_seqvlad_with_final_norm_Me_18854906430123
// MI455X (gfx1250) — compile-verified, matrix operand order checked by reading, not run on hardware
//
#include <hip/hip_runtime.h>
#include <hip/hip_bf16.h>

typedef __attribute__((ext_vector_type(16))) _Float16 v16h;
typedef __attribute__((ext_vector_type(8)))  _Float16 v8h;
typedef __attribute__((ext_vector_type(2)))  _Float16 v2h;
typedef __attribute__((ext_vector_type(8)))  float    v8f;

#define NBATCH 32
#define SEQ    5
#define CDIM   768
#define PDIM   1280
#define KCLUST 64
#define HWSZ   256
#define EPSN   1e-12f
#define LPITCH 40   // LDS row pitch in halves: 80B = 20-word bank stride, 16B aligned

// ---- workspace layout (bytes) ----
#define OFF_BSM   0u          // [32][64][1280] f16
#define OFF_ASUM  5242880u    // [32][64] f32
#define OFF_VLAD  5251072u    // [32][64][768] f32
#define OFF_SSUM  11542528u   // [32][64] f32
#define OFF_SCALE 11550720u   // [32][64] f32

// Swap bits 3<->4 of the K index: row stored as [K0..7,K16..23,K8..15,K24..31]
// so each WMMA lane-half reads 16 contiguous halves (2x ds_load_b128).
__device__ __forceinline__ int kswz(int c) {
  return (c & ~24) | ((c & 8) << 1) | ((c & 16) >> 1);
}

__device__ __forceinline__ v16h frag16(const _Float16* p) {
  v8h lo = *(const v8h*)p;
  v8h hi = *(const v8h*)(p + 8);
  return __builtin_shufflevector(lo, hi, 0, 1, 2, 3, 4, 5, 6, 7,
                                 8, 9, 10, 11, 12, 13, 14, 15);
}

__global__ void seqvlad_zero_kernel(float* __restrict__ asum,
                                    float* __restrict__ ssum) {
  int i = blockIdx.x * 256 + threadIdx.x;
  if (i < NBATCH * KCLUST) { asum[i] = 0.f; ssum[i] = 0.f; }
}

// ---------------------------------------------------------------------------
// Kernel A: per batch, P-tile of 128 columns.
//   logits = conv_w(64x768) x frames(768x128) via v_wmma_f32_16x16x32_f16,
//   column sumsq fused into staging, softmax over 64 clusters,
//   emits bsm = softmax*inv (f16) and accumulates asum.
// ---------------------------------------------------------------------------
__global__ __launch_bounds__(256) void seqvlad_assign_kernel(
    const float* __restrict__ frames, const float* __restrict__ conv_w,
    _Float16* __restrict__ bsm, float* __restrict__ asum) {
  __shared__ _Float16 Wt[KCLUST][LPITCH] __attribute__((aligned(16)));
  __shared__ _Float16 Xt[128][LPITCH] __attribute__((aligned(16)));
  __shared__ float    Lbuf[KCLUST][128];
  __shared__ float    sA[KCLUST];
  __shared__ float    colsq[128];

  const int tid = threadIdx.x;
  const int b   = blockIdx.x / 10;
  const int p0  = (blockIdx.x % 10) * 128;
  const int s   = p0 / HWSZ, hw0 = p0 % HWSZ;
  if (tid < KCLUST) sA[tid] = 0.f;

  const int w = tid >> 5, lane = tid & 31;
  const int mb  = (w & 3) * 16;            // cluster-row tile base
  const int nbb = (w >> 2) * 64;           // column tile base
  const int rowsel = lane & 15;
  const int kh     = (lane & 16) ? 16 : 0; // contiguous fragment offset (halves)
  const int rowoff = (lane & 16) ? 8 : 0;  // C/D row offset per ISA layout

  v8f acc[4] = {};
  float csq = 0.f;  // partial column sum-of-squares (fixed j = tid&127)

  const float* fbase = frames + ((size_t)(b * SEQ + s) * CDIM) * HWSZ + hw0;

  for (int kk = 0; kk < CDIM; kk += 32) {
    // W tile: 64x32, 2 halves packed per ds_store_b32
    for (int e2 = tid; e2 < KCLUST * 16; e2 += 256) {
      int k = e2 >> 4, c2 = (e2 & 15) * 2;
      float2 v = *(const float2*)&conv_w[k * CDIM + kk + c2];
      *(v2h*)&Wt[k][kswz(c2)] = v2h{(_Float16)v.x, (_Float16)v.y};
    }
    // X tile: 32x128, stored transposed fragment-ready [col][kswz(K)]
    for (int e2 = tid; e2 < 16 * 128; e2 += 256) {
      int j = e2 & 127, c2 = (e2 >> 7) * 2;
      float v0 = fbase[(size_t)(kk + c2) * HWSZ + j];
      float v1 = fbase[(size_t)(kk + c2 + 1) * HWSZ + j];
      csq = fmaf(v0, v0, fmaf(v1, v1, csq));  // fused descriptor-norm pass
      *(v2h*)&Xt[j][kswz(c2)] = v2h{(_Float16)v0, (_Float16)v1};
    }
    __syncthreads();
    if (kk + 32 < CDIM)
      __builtin_prefetch(&fbase[(size_t)(kk + 32 + (tid >> 7)) * HWSZ + (tid & 127)], 0, 1);

    const v16h a = frag16(&Wt[mb + rowsel][kh]);
#pragma unroll
    for (int nt = 0; nt < 4; ++nt) {
      const v16h bf = frag16(&Xt[nbb + nt * 16 + rowsel][kh]);
      acc[nt] = __builtin_amdgcn_wmma_f32_16x16x32_f16(
          false, a, false, bf, (short)0, acc[nt], false, false);
    }
    __syncthreads();
  }

  // combine per-column sumsq halves and spill logits
  if (tid >= 128) colsq[tid - 128] = csq;
#pragma unroll
  for (int nt = 0; nt < 4; ++nt) {
    const int col = nbb + nt * 16 + rowsel;
#pragma unroll
    for (int rr = 0; rr < 8; ++rr) Lbuf[mb + rowoff + rr][col] = acc[nt][rr];
  }
  __syncthreads();

  if (tid < 128) {
    const int j = tid, p = p0 + j;
    const float inv = 1.f / fmaxf(sqrtf(csq + colsq[j]), EPSN);
    float m = -3.0e38f;
#pragma unroll 4
    for (int k = 0; k < KCLUST; ++k) m = fmaxf(m, Lbuf[k][j] * inv);
    float se = 0.f;
#pragma unroll 4
    for (int k = 0; k < KCLUST; ++k) {
      float e = __expf(Lbuf[k][j] * inv - m);
      Lbuf[k][j] = e;
      se += e;
    }
    const float rs = 1.f / se;
#pragma unroll 4
    for (int k = 0; k < KCLUST; ++k) {
      float ap = Lbuf[k][j] * rs;
      atomicAdd(&sA[k], ap);
      bsm[((size_t)b * KCLUST + k) * PDIM + p] = (_Float16)(ap * inv);
    }
  }
  __syncthreads();
  if (tid < KCLUST) atomicAdd(&asum[b * KCLUST + tid], sA[tid]);
}

// ---------------------------------------------------------------------------
// Kernel B: per batch, C-tile of 128.
//   vlad = bsm(64x1280) x framesT(1280x128) via v_wmma_f32_16x16x32_f16,
//   fused centroid subtraction + per-(n,k) sum-of-squares.
// ---------------------------------------------------------------------------
__global__ __launch_bounds__(256) void seqvlad_vlad_kernel(
    const float* __restrict__ frames, const _Float16* __restrict__ bsm,
    const float* __restrict__ asum, const float* __restrict__ centroids,
    float* __restrict__ vlad, float* __restrict__ ssum) {
  __shared__ _Float16 At[KCLUST][LPITCH] __attribute__((aligned(16)));
  __shared__ _Float16 Bt[128][LPITCH] __attribute__((aligned(16)));
  __shared__ float    sS[KCLUST];

  const int tid = threadIdx.x;
  const int b  = blockIdx.x / 6;
  const int c0 = (blockIdx.x % 6) * 128;
  if (tid < KCLUST) sS[tid] = 0.f;

  const int w = tid >> 5, lane = tid & 31;
  const int mb = (w & 3) * 16, nbb = (w >> 2) * 64;
  const int rowsel = lane & 15;
  const int kh     = (lane & 16) ? 16 : 0;
  const int rowoff = (lane & 16) ? 8 : 0;

  v8f acc[4] = {};

  for (int pp = 0; pp < PDIM; pp += 32) {
    const int s = pp / HWSZ, hw0 = pp % HWSZ;
    const float* fbase = frames + ((size_t)(b * SEQ + s) * CDIM + c0) * HWSZ + hw0;
    // A tile: bsm 64x32 f16, copy 2 halves per b32
    for (int e2 = tid; e2 < KCLUST * 16; e2 += 256) {
      int k = e2 >> 4, j2 = (e2 & 15) * 2;
      v2h t = *(const v2h*)&bsm[((size_t)b * KCLUST + k) * PDIM + pp + j2];
      *(v2h*)&At[k][kswz(j2)] = t;
    }
    // B tile: frames [c][p] -> fragment-ready [c][kswz(p')]
    for (int e2 = tid; e2 < 128 * 16; e2 += 256) {
      int cc = e2 >> 4, j2 = (e2 & 15) * 2;
      float2 v = *(const float2*)&fbase[(size_t)cc * HWSZ + j2];
      *(v2h*)&Bt[cc][kswz(j2)] = v2h{(_Float16)v.x, (_Float16)v.y};
    }
    __syncthreads();
    if (pp + 32 < PDIM)
      __builtin_prefetch(&fbase[(size_t)(tid >> 1) * HWSZ + ((tid & 1) * 16) + 32], 0, 1);

    const v16h a = frag16(&At[mb + rowsel][kh]);
#pragma unroll
    for (int nt = 0; nt < 4; ++nt) {
      const v16h bf = frag16(&Bt[nbb + nt * 16 + rowsel][kh]);
      acc[nt] = __builtin_amdgcn_wmma_f32_16x16x32_f16(
          false, a, false, bf, (short)0, acc[nt], false, false);
    }
    __syncthreads();
  }

  float ssl[8] = {0.f, 0.f, 0.f, 0.f, 0.f, 0.f, 0.f, 0.f};
#pragma unroll
  for (int nt = 0; nt < 4; ++nt) {
    const int c = c0 + nbb + nt * 16 + rowsel;
#pragma unroll
    for (int rr = 0; rr < 8; ++rr) {
      const int k = mb + rowoff + rr;
      float v = acc[nt][rr] - asum[b * KCLUST + k] * centroids[k * CDIM + c];
      vlad[((size_t)b * KCLUST + k) * CDIM + c] = v;
      ssl[rr] = fmaf(v, v, ssl[rr]);
    }
  }
#pragma unroll
  for (int rr = 0; rr < 8; ++rr) atomicAdd(&sS[mb + rowoff + rr], ssl[rr]);
  __syncthreads();
  if (tid < KCLUST) atomicAdd(&ssum[b * KCLUST + tid], sS[tid]);
}

// scale[n][k] = (1/max(||v_k||,eps)) * (1/max(||vlad_n||_after_intra, eps))
__global__ void seqvlad_scale_kernel(const float* __restrict__ ssum,
                                     float* __restrict__ scale) {
  __shared__ float red[KCLUST];
  const int n = blockIdx.x, t = threadIdx.x;
  const float ss  = ssum[n * KCLUST + t];
  const float rin = 1.f / fmaxf(sqrtf(ss), EPSN);
  red[t] = ss * rin * rin;
  __syncthreads();
  for (int off = 32; off > 0; off >>= 1) {
    if (t < off) red[t] += red[t + off];
    __syncthreads();
  }
  const float fin = 1.f / fmaxf(sqrtf(red[0]), EPSN);
  scale[n * KCLUST + t] = rin * fin;
}

__global__ void seqvlad_out_kernel(const float* __restrict__ vlad,
                                   const float* __restrict__ scale,
                                   float* __restrict__ out) {
  const int i4 = blockIdx.x * 256 + threadIdx.x;  // float4 index
  const int base = i4 * 4;
  const int n = base / (KCLUST * CDIM);
  const int k = (base / CDIM) % KCLUST;
  float4 v = ((const float4*)vlad)[i4];
  const float sc = scale[n * KCLUST + k];
  v.x *= sc; v.y *= sc; v.z *= sc; v.w *= sc;
  ((float4*)out)[i4] = v;
}

extern "C" void kernel_launch(void* const* d_in, const int* in_sizes, int n_in,
                              void* d_out, int out_size, void* d_ws, size_t ws_size,
                              hipStream_t stream) {
  (void)in_sizes; (void)n_in; (void)out_size; (void)ws_size;
  const float* frames    = (const float*)d_in[0];
  const float* conv_w    = (const float*)d_in[1];
  const float* centroids = (const float*)d_in[2];
  char* ws = (char*)d_ws;
  _Float16* bsm = (_Float16*)(ws + OFF_BSM);
  float* asum   = (float*)(ws + OFF_ASUM);
  float* vlad   = (float*)(ws + OFF_VLAD);
  float* ssum   = (float*)(ws + OFF_SSUM);
  float* scale  = (float*)(ws + OFF_SCALE);
  float* out    = (float*)d_out;

  seqvlad_zero_kernel<<<dim3(8), dim3(256), 0, stream>>>(asum, ssum);
  seqvlad_assign_kernel<<<dim3(NBATCH * 10), dim3(256), 0, stream>>>(frames, conv_w, bsm, asum);
  seqvlad_vlad_kernel<<<dim3(NBATCH * 6), dim3(256), 0, stream>>>(frames, bsm, asum, centroids, vlad, ssum);
  seqvlad_scale_kernel<<<dim3(NBATCH), dim3(KCLUST), 0, stream>>>(ssum, scale);
  seqvlad_out_kernel<<<dim3(1536), dim3(256), 0, stream>>>(vlad, scale, out);
}
